// Nsct_swinTransformer_25984552141021
// MI455X (gfx1250) — matrix operand order repaired and checked by reading, not compiled, not run on hardware
//
// MI455X (gfx1250, CDNA5) NSCT + Swin implementation.
// - All dense GEMMs (QKV/proj/fc1/fc2/lin) on v_wmma_f32_16x16x32_bf16,
//   128x128x32 tiles, double-buffered LDS staged via GLOBAL_LOAD_ASYNC_TO_LDS
//   (ASYNCcnt / s_wait_asynccnt), fused bias/GELU/residual/window-scatter.
// - Window attention (49 tok, 8 heads) also on WMMA: QK^T padded 64x64x96,
//   P*V padded 64x80x64; only the softmax stays on f32 VALU.
// - NSCT 5x5 banks / LN / final ICT+1x1-conv stay f32 VALU (tiny FLOPs).
#include <hip/hip_runtime.h>
#include <hip/hip_bf16.h>
#include <stdint.h>

typedef __bf16 bf16_t;
typedef __attribute__((ext_vector_type(16))) __bf16 bfx16;
typedef __attribute__((ext_vector_type(8)))  __bf16 bfx8;
typedef __attribute__((ext_vector_type(8)))  float  fx8;

#define DEV static __device__ __forceinline__

constexpr int kB = 8, kC = 64, kH = 56, kW = 56;
constexpr int kLM = 9, kD = 640, kWin = 7, kHeads = 8, kDh = 80;
constexpr int kTok = kB * kH * kW;           // 25088 = 196 * 128
constexpr int kWins = kB * (kH / kWin) * (kW / kWin); // 512
constexpr float kPi = 3.14159265358979323846f;

DEV float geluf(float x) {
    float x3 = x * x * x;
    return 0.5f * x * (1.0f + tanhf(0.7978845608028654f * (x + 0.044715f * x3)));
}

DEV int rid_sw(int p) { return p < 49 ? 0 : (p < 53 ? 1 : 2); }

// windowed gemm-row -> spatial token row, applying inverse cyclic shift
DEV int remap_row(int g, int shift) {
    int win = g / 49, t = g % 49;
    int b = win >> 6, wr = win & 63;
    int wh = wr >> 3, ww = wr & 7;
    int i = t / 7, j = t % 7;
    int h = wh * 7 + i + shift; if (h >= 56) h -= 56;
    int w = ww * 7 + j + shift; if (w >= 56) w -= 56;
    return (b * 56 + h) * 56 + w;
}

// ---- CDNA5 async global->LDS staging (ASYNCcnt) -------------------------------
DEV void async_ld16(unsigned lds_off, const void* gaddr) {
    asm volatile("global_load_async_to_lds_b128 %0, %1, off"
                 :: "v"(lds_off), "v"(gaddr) : "memory");
}
DEV void wait_async0() {
    asm volatile("s_wait_asynccnt 0x0" ::: "memory");
}

// ---- WMMA fragment loads from LDS (ISA 7.12.2 wave32 layouts) -----------------
// A 16x32 bf16 tile at `base` (row-major, element stride `stride`):
//   lane<16 -> M=lane, K {0..7,16..23}; lane>=16 -> M=lane-16, K {8..15,24..31}
DEV bfx16 ld_afrag(const bf16_t* base, int stride, int lane) {
    const bf16_t* p = base + (lane & 15) * stride + ((lane >> 4) << 3);
    bfx8 lo = *(const bfx8*)p;
    bfx8 hi = *(const bfx8*)(p + 16);
    bfx16 r;
#pragma unroll
    for (int e = 0; e < 8; ++e) { r[e] = lo[e]; r[e + 8] = hi[e]; }
    return r;
}
// B 32x16 bf16 tile staged column-major: Bs[col][k] with element stride `stride`:
//   lane<16 -> N=lane, K {0..7,16..23}; lane>=16 -> N=lane-16, K {8..15,24..31}
DEV bfx16 ld_bfrag(const bf16_t* base, int stride, int lane) {
    const bf16_t* p = base + (lane & 15) * stride + ((lane >> 4) << 3);
    bfx8 lo = *(const bfx8*)p;
    bfx8 hi = *(const bfx8*)(p + 16);
    bfx16 r;
#pragma unroll
    for (int e = 0; e < 8; ++e) { r[e] = lo[e]; r[e + 8] = hi[e]; }
    return r;
}

// Fixed 9-band NSCT filter bank (lowpass + 8 directional), matches reference.
__device__ void gen_filters(float* F) {
    float g[25]; float gs = 0.f;
    for (int p = 0; p < 25; ++p) {
        int dy = p / 5 - 2, dx = p % 5 - 2;
        g[p] = expf(-(float)(dx * dx + dy * dy) / 4.0f);
        gs += g[p];
    }
    for (int p = 0; p < 25; ++p) { g[p] /= gs; F[p] = g[p]; }
    for (int k = 0; k < 8; ++k) {
        float th = (float)k * kPi / 8.0f;
        float c = cosf(th), s = sinf(th);
        float d[25]; float mn = 0.f;
        for (int p = 0; p < 25; ++p) {
            int dy = p / 5 - 2, dx = p % 5 - 2;
            d[p] = (c * (float)dx + s * (float)dy) * g[p];
            mn += d[p];
        }
        mn /= 25.f;
        float n2 = 0.f;
        for (int p = 0; p < 25; ++p) { d[p] -= mn; n2 += d[p] * d[p]; }
        float inv = rsqrtf(n2);
        for (int p = 0; p < 25; ++p) F[(k + 1) * 25 + p] = d[p] * inv;
    }
}

// ---------------- weight transpose + cast: f32 [K,N] -> bf16 [N][K] ------------
__global__ __launch_bounds__(256) void wtrans(const float* __restrict__ in,
                                              bf16_t* __restrict__ out,
                                              int K, int N) {
    int idx = blockIdx.x * 256 + threadIdx.x;
    if (idx < N * K) {
        int n = idx / K, k = idx % K;
        out[idx] = (bf16_t)in[(size_t)k * N + n];
    }
}

__global__ __launch_bounds__(256) void cast_f32_bf16(const float* __restrict__ in,
                                                     bf16_t* __restrict__ out,
                                                     size_t n) {
    size_t i = (size_t)blockIdx.x * 256 + threadIdx.x;
    if (i < n) out[i] = (bf16_t)in[i];
}

// ---------------- fused ReLU + NSCT 9-band + concat -> BHWC f32 ----------------
__global__ __launch_bounds__(256) void nsct_fwd(const float* __restrict__ x,
                                                float* __restrict__ x3) {
    __shared__ float tile[60][60];
    __shared__ float F[225];
    int bc = blockIdx.x;
    int b = bc >> 6, c = bc & 63;
    const float* xp = x + ((size_t)b * 64 + c) * 3136;
    for (int p = threadIdx.x; p < 3600; p += 256) {
        int yy = p / 60 - 2, xx = p % 60 - 2;
        float v = 0.f;
        if (yy >= 0 && yy < 56 && xx >= 0 && xx < 56) {
            v = xp[yy * 56 + xx];
            v = v > 0.f ? v : 0.f;
        }
        tile[p / 60][p % 60] = v;
    }
    if (threadIdx.x == 0) gen_filters(F);
    __syncthreads();
    for (int p = threadIdx.x; p < 3136; p += 256) {
        int h = p / 56, w = p % 56;
        float s[9];
#pragma unroll
        for (int bd = 0; bd < 9; ++bd) s[bd] = 0.f;
#pragma unroll
        for (int dy = 0; dy < 5; ++dy)
#pragma unroll
            for (int dx = 0; dx < 5; ++dx) {
                float v = tile[h + dy][w + dx];
#pragma unroll
                for (int bd = 0; bd < 9; ++bd) s[bd] += F[bd * 25 + dy * 5 + dx] * v;
            }
        float* orow = x3 + ((size_t)(b * 56 + h) * 56 + w) * 640;
#pragma unroll
        for (int bd = 0; bd < 9; ++bd) orow[bd * 64 + c] = s[bd];
        orow[576 + c] = tile[h + 2][w + 2];
    }
}

// ---------------- LayerNorm (optionally shift+window-partition), f32 -> bf16 ---
template <bool WINDOWED>
__global__ __launch_bounds__(256) void ln_kernel(const float* __restrict__ x,
                                                 const float* __restrict__ g,
                                                 const float* __restrict__ b,
                                                 bf16_t* __restrict__ out,
                                                 int shift) {
    __shared__ float red[256], red2[256];
    __shared__ float s_mean, s_rstd;
    int row = blockIdx.x;
    int src;
    if (WINDOWED) {
        int win = row / 49, t = row % 49;
        int bb = win >> 6, wr = win & 63, wh = wr >> 3, ww = wr & 7;
        int i = t / 7, j = t % 7;
        int h = wh * 7 + i + shift; if (h >= 56) h -= 56;
        int w = ww * 7 + j + shift; if (w >= 56) w -= 56;
        src = (bb * 56 + h) * 56 + w;
    } else {
        src = row;
    }
    const float* xr = x + (size_t)src * 640;
    float s = 0.f, s2 = 0.f;
    for (int d = threadIdx.x; d < 640; d += 256) {
        float v = xr[d];
        s += v; s2 += v * v;
    }
    red[threadIdx.x] = s; red2[threadIdx.x] = s2;
    __syncthreads();
    for (int off = 128; off > 0; off >>= 1) {
        if ((int)threadIdx.x < off) {
            red[threadIdx.x] += red[threadIdx.x + off];
            red2[threadIdx.x] += red2[threadIdx.x + off];
        }
        __syncthreads();
    }
    if (threadIdx.x == 0) {
        float m = red[0] / 640.f;
        float var = red2[0] / 640.f - m * m;
        s_mean = m;
        s_rstd = rsqrtf(var + 1e-5f);
    }
    __syncthreads();
    float m = s_mean, rs = s_rstd;
    bf16_t* orow = out + (size_t)row * 640;
    for (int d = threadIdx.x; d < 640; d += 256)
        orow[d] = (bf16_t)((xr[d] - m) * rs * g[d] + b[d]);
}

// ---------------- WMMA bf16 GEMM: C[M,N] = A[M,K] * Bt[N,K]^T + bias ----------
// OUTMODE: 0 = bf16 out, 1 = f32 out, 2 = f32 residual accumulate (+= into outF)
// ACT:     0 = none, 1 = GELU(tanh)
// REMAP:   windowed-row -> spatial-row scatter with inverse shift (proj epilogue)
template <int OUTMODE, int ACT, bool REMAP>
__global__ __launch_bounds__(256) void gemm_bf16_wmma(
    const bf16_t* __restrict__ A, const bf16_t* __restrict__ Bt,
    const float* __restrict__ bias,
    float* __restrict__ outF, bf16_t* __restrict__ outB,
    int M, int N, int K, int shift) {
    __shared__ bf16_t As[2][128][40];   // [row][k], padded stride
    __shared__ bf16_t Bs[2][128][40];   // [col][k] (B pre-transposed in global)

    const int tid = threadIdx.x;
    const int lane = tid & 31;
    const int wid = tid >> 5;
    const int wm = wid >> 2;   // 0..1  (64 rows per wave)
    const int wn = wid & 3;    // 0..3  (32 cols per wave)
    const int m0 = blockIdx.y * 128;
    const int n0 = blockIdx.x * 128;

    fx8 acc[4][2];
#pragma unroll
    for (int i = 0; i < 4; ++i)
#pragma unroll
        for (int j = 0; j < 2; ++j)
#pragma unroll
            for (int e = 0; e < 8; ++e) acc[i][j][e] = 0.0f;

    const int nk = K >> 5;
    // staging: 512 16B segments per tile per matrix; 2 per thread, async to LDS
    const int r0 = tid >> 2, sg = tid & 3;
    const size_t aOff0 = (size_t)(m0 + r0) * K + sg * 8;
    const size_t aOff1 = (size_t)(m0 + r0 + 64) * K + sg * 8;
    const bool aV0 = (m0 + r0) < M, aV1 = (m0 + r0 + 64) < M;
    const size_t bOff0 = (size_t)(n0 + r0) * K + sg * 8;
    const size_t bOff1 = (size_t)(n0 + r0 + 64) * K + sg * 8;
    const bool bV0 = (n0 + r0) < N, bV1 = (n0 + r0 + 64) < N;

    const unsigned ldsA0 = (unsigned)(uintptr_t)&As[0][r0][sg * 8];
    const unsigned ldsA1 = (unsigned)(uintptr_t)&As[0][r0 + 64][sg * 8];
    const unsigned ldsB0 = (unsigned)(uintptr_t)&Bs[0][r0][sg * 8];
    const unsigned ldsB1 = (unsigned)(uintptr_t)&Bs[0][r0 + 64][sg * 8];
    const unsigned bufStride = 128u * 40u * (unsigned)sizeof(bf16_t);

    auto issue = [&](int kt, int buf) {
        size_t ko = (size_t)kt * 32;
        unsigned bo = (unsigned)buf * bufStride;
        if (aV0) async_ld16(ldsA0 + bo, A + aOff0 + ko);
        if (aV1) async_ld16(ldsA1 + bo, A + aOff1 + ko);
        if (bV0) async_ld16(ldsB0 + bo, Bt + bOff0 + ko);
        if (bV1) async_ld16(ldsB1 + bo, Bt + bOff1 + ko);
    };

    auto compute = [&](int buf) {
        bfx16 af[4], bfr[2];
#pragma unroll
        for (int mi = 0; mi < 4; ++mi)
            af[mi] = ld_afrag(&As[buf][wm * 64 + mi * 16][0], 40, lane);
#pragma unroll
        for (int ni = 0; ni < 2; ++ni)
            bfr[ni] = ld_bfrag(&Bs[buf][wn * 32 + ni * 16][0], 40, lane);
#pragma unroll
        for (int mi = 0; mi < 4; ++mi)
#pragma unroll
            for (int ni = 0; ni < 2; ++ni)
                acc[mi][ni] = __builtin_amdgcn_wmma_f32_16x16x32_bf16(
                    false, af[mi], false, bfr[ni], (short)0, acc[mi][ni], false, false);
    };

    issue(0, 0);
    wait_async0();
    __syncthreads();
    for (int kt = 0; kt < nk; ++kt) {
        const int cur = kt & 1;
        if (kt + 1 < nk) issue(kt + 1, cur ^ 1);  // overlaps with compute below
        compute(cur);
        wait_async0();
        __syncthreads();
    }

    // Epilogue.  C layout: VGPR r -> row = r + 8*(lane>=16), col = lane&15.
    const int crow0 = wm * 64 + ((lane >> 4) << 3);
    const int ccol = wn * 32 + (lane & 15);
#pragma unroll
    for (int mi = 0; mi < 4; ++mi)
#pragma unroll
        for (int ni = 0; ni < 2; ++ni) {
            int gcol = n0 + ccol + ni * 16;
            if (gcol >= N) continue;
            float bv = bias ? bias[gcol] : 0.f;
#pragma unroll
            for (int r = 0; r < 8; ++r) {
                int grow = m0 + crow0 + mi * 16 + r;
                if (grow >= M) continue;
                float v = acc[mi][ni][r] + bv;
                if (ACT == 1) v = geluf(v);
                if (OUTMODE == 0) {
                    outB[(size_t)grow * N + gcol] = (bf16_t)v;
                } else if (OUTMODE == 1) {
                    outF[(size_t)grow * N + gcol] = v;
                } else {
                    size_t drow = REMAP ? (size_t)remap_row(grow, shift) : (size_t)grow;
                    outF[drow * N + gcol] += v;
                }
            }
        }
}

// ---------------- window attention on WMMA -------------------------------------
// One block = (window, head), 128 threads = 4 waves.
// S = Q*K^T : 64x64x96 (49 tokens zero-padded to 64, dh 80 zero-padded to 96)
// O = P*V   : 64x80x64
__global__ __launch_bounds__(128) void attn_wmma(const bf16_t* __restrict__ qkv,
                                                 bf16_t* __restrict__ aout,
                                                 int shift) {
    __shared__ bf16_t qa[64 * 96];   // A operand: Q, K-dim padded to 96
    __shared__ bf16_t kb[64 * 96];   // B staged [col=token u][k=d]
    __shared__ bf16_t vt[80 * 64];   // B staged [col=d][k=token u]
    __shared__ bf16_t pa[64 * 72];   // A operand: probabilities, k=u padded
    __shared__ float  sS[64 * 68];   // f32 scores

    const int tid = threadIdx.x;
    const int lane = tid & 31;
    const int wv = tid >> 5;       // 0..3, one 16-row M-tile per wave
    const int win = blockIdx.x, head = blockIdx.y;
    const int wr = win & 63, wh = wr >> 3, ww = wr & 7;
    const bf16_t* base = qkv + (size_t)win * 49 * 1920 + head * 80;

    // stage Q (pre-scaled), K, V^T with zero padding
    for (int idx = tid; idx < 64 * 96; idx += 128) {
        int t = idx / 96, d = idx % 96;
        float qv = 0.f, kv = 0.f;
        if (t < 49 && d < 80) {
            const bf16_t* p = base + (size_t)t * 1920 + d;
            qv = (float)p[0] * 0.11180339887498949f;   // 1/sqrt(80)
            kv = (float)p[640];
        }
        qa[idx] = (bf16_t)qv;
        kb[idx] = (bf16_t)kv;
    }
    for (int idx = tid; idx < 80 * 64; idx += 128) {
        int d = idx / 64, u = idx % 64;
        float vv = 0.f;
        if (u < 49) vv = (float)base[(size_t)u * 1920 + 1280 + d];
        vt[idx] = (bf16_t)vv;
    }
    __syncthreads();

    // S = Q * K^T   (per wave: 1 M-tile x 4 N-tiles x 3 K-steps)
    fx8 accS[4];
#pragma unroll
    for (int n = 0; n < 4; ++n)
#pragma unroll
        for (int e = 0; e < 8; ++e) accS[n][e] = 0.f;
#pragma unroll
    for (int kt = 0; kt < 96; kt += 32) {
        bfx16 af = ld_afrag(qa + (wv * 16) * 96 + kt, 96, lane);
#pragma unroll
        for (int n = 0; n < 4; ++n) {
            bfx16 bfr = ld_bfrag(kb + (n * 16) * 96 + kt, 96, lane);
            accS[n] = __builtin_amdgcn_wmma_f32_16x16x32_bf16(
                false, af, false, bfr, (short)0, accS[n], false, false);
        }
    }
    const int rbase = wv * 16 + ((lane >> 4) << 3);
    const int cbase = lane & 15;
#pragma unroll
    for (int n = 0; n < 4; ++n)
#pragma unroll
        for (int r = 0; r < 8; ++r)
            sS[(rbase + r) * 68 + n * 16 + cbase] = accS[n][r];
    __syncthreads();

    // masked softmax (rows) -> bf16 probability matrix (A operand, zero padded)
    if (tid < 64) {
        int t = tid;
        if (t < 49) {
            int rt = 0;
            if (shift) rt = rid_sw(wh * 7 + t / 7) * 3 + rid_sw(ww * 7 + t % 7);
            float mx = -3e38f;
            for (int u = 0; u < 49; ++u) {
                float v = sS[t * 68 + u];
                if (shift) {
                    int ru = rid_sw(wh * 7 + u / 7) * 3 + rid_sw(ww * 7 + u % 7);
                    if (ru != rt) v = -1e9f;
                }
                sS[t * 68 + u] = v;
                mx = fmaxf(mx, v);
            }
            float sm = 0.f;
            for (int u = 0; u < 49; ++u) {
                float e = expf(sS[t * 68 + u] - mx);
                sS[t * 68 + u] = e;
                sm += e;
            }
            float inv = 1.f / sm;
            for (int u = 0; u < 64; ++u)
                pa[t * 72 + u] = (bf16_t)((u < 49) ? sS[t * 68 + u] * inv : 0.f);
        } else {
            for (int u = 0; u < 64; ++u) pa[t * 72 + u] = (bf16_t)0.f;
        }
    }
    __syncthreads();

    // O = P * V    (per wave: 1 M-tile x 5 N-tiles x 2 K-steps)
    fx8 accO[5];
#pragma unroll
    for (int n = 0; n < 5; ++n)
#pragma unroll
        for (int e = 0; e < 8; ++e) accO[n][e] = 0.f;
#pragma unroll
    for (int kt = 0; kt < 64; kt += 32) {
        bfx16 af = ld_afrag(pa + (wv * 16) * 72 + kt, 72, lane);
#pragma unroll
        for (int n = 0; n < 5; ++n) {
            bfx16 bfr = ld_bfrag(vt + (n * 16) * 64 + kt, 64, lane);
            accO[n] = __builtin_amdgcn_wmma_f32_16x16x32_bf16(
                false, af, false, bfr, (short)0, accO[n], false, false);
        }
    }
#pragma unroll
    for (int n = 0; n < 5; ++n)
#pragma unroll
        for (int r = 0; r < 8; ++r) {
            int t = rbase + r, d = n * 16 + cbase;
            if (t < 49 && d < 80)
                aout[((size_t)win * 49 + t) * 640 + head * 80 + d] = (bf16_t)accO[n][r];
        }
}

// ---------------- fused ICT (9-band inverse) + concat + 1x1 conv ---------------
__global__ __launch_bounds__(256) void ict_out_kernel(const float* __restrict__ y,
                                                      const float* __restrict__ cw,
                                                      const float* __restrict__ cb,
                                                      float* __restrict__ out) {
    __shared__ float yn[25][576];
    __shared__ float F[225];
    __shared__ float ytr[64];
    int pix = blockIdx.x;
    int b = pix / 3136, rem = pix % 3136, h = rem / 56, w = rem % 56;
    for (int idx = threadIdx.x; idx < 14400; idx += 256) {
        int qd = idx / 576, j = idx % 576;
        int dy = qd / 5 - 2, dx = qd % 5 - 2;
        int hh = h + dy, ww2 = w + dx;
        float vv = 0.f;
        if (hh >= 0 && hh < 56 && ww2 >= 0 && ww2 < 56)
            vv = y[((size_t)(b * 56 + hh) * 56 + ww2) * 576 + j];
        yn[qd][j] = vv;
    }
    if (threadIdx.x == 0) gen_filters(F);
    __syncthreads();
    if (threadIdx.x < 64) {
        int c = threadIdx.x;
        float s = 0.f;
        for (int bd = 0; bd < 9; ++bd)
            for (int p = 0; p < 25; ++p) s += F[bd * 25 + p] * yn[p][bd * 64 + c];
        ytr[c] = s;
    }
    __syncthreads();
    if (threadIdx.x < 64) {
        int o = threadIdx.x;
        const float* wrow = cw + o * 640;
        float s = cb[o];
        for (int c = 0; c < 64; ++c) s += wrow[c] * ytr[c];
        const float* yc = &yn[12][0];   // center pixel (dy=dx=0)
        for (int j = 0; j < 576; ++j) s += wrow[64 + j] * yc[j];
        out[((size_t)(b * 64 + o) * 56 + h) * 56 + w] = s;
    }
}

// -------------------------------- launcher ------------------------------------
extern "C" void kernel_launch(void* const* d_in, const int* in_sizes, int n_in,
                              void* d_out, int out_size, void* d_ws, size_t ws_size,
                              hipStream_t stream) {
    const float* x      = (const float*)d_in[0];
    const float* ln1_g  = (const float*)d_in[1];
    const float* ln1_b  = (const float*)d_in[2];
    const float* qkv_w  = (const float*)d_in[3];
    const float* qkv_b  = (const float*)d_in[4];
    const float* proj_w = (const float*)d_in[5];
    const float* proj_b = (const float*)d_in[6];
    const float* ln2_g  = (const float*)d_in[7];
    const float* ln2_b  = (const float*)d_in[8];
    const float* fc1_w  = (const float*)d_in[9];
    const float* fc1_b  = (const float*)d_in[10];
    const float* fc2_w  = (const float*)d_in[11];
    const float* fc2_b  = (const float*)d_in[12];
    const float* lin_w  = (const float*)d_in[13];
    const float* lin_b  = (const float*)d_in[14];
    const float* conv_w = (const float*)d_in[15];
    const float* conv_b = (const float*)d_in[16];

    char* ws = (char*)d_ws;
    size_t off = 0;
    auto alloc = [&](size_t bytes) {
        size_t o = off;
        off += (bytes + 255) & ~(size_t)255;
        return o;
    };
    float*  x3    = (float*)(ws + alloc((size_t)kTok * 640 * 4));
    bf16_t* bufA  = (bf16_t*)(ws + alloc((size_t)kTok * 640 * 2));   // LN out / attn out
    char*   qkvB  = ws + alloc((size_t)kTok * 1920 * 2);             // qkv bf16, reused as y f32
    bf16_t* qkv   = (bf16_t*)qkvB;
    float*  ybuf  = (float*)qkvB;
    bf16_t* ff    = (bf16_t*)(ws + alloc((size_t)kTok * 2560 * 2));
    bf16_t* qkvwt = (bf16_t*)(ws + alloc((size_t)2 * 1920 * 640 * 2));
    bf16_t* projwt= (bf16_t*)(ws + alloc((size_t)2 * 640 * 640 * 2));
    bf16_t* fc1wt = (bf16_t*)(ws + alloc((size_t)2 * 2560 * 640 * 2));
    bf16_t* fc2wt = (bf16_t*)(ws + alloc((size_t)2 * 640 * 2560 * 2));
    bf16_t* linwt = (bf16_t*)(ws + alloc((size_t)576 * 640 * 2));

    auto wt = [&](const float* src, bf16_t* dst, int K, int N) {
        int n = N * K;
        wtrans<<<(n + 255) / 256, 256, 0, stream>>>(src, dst, K, N);
    };
    for (int l = 0; l < 2; ++l) {
        wt(qkv_w + (size_t)l * 640 * 1920, qkvwt + (size_t)l * 1920 * 640, 640, 1920);
        wt(proj_w + (size_t)l * 640 * 640, projwt + (size_t)l * 640 * 640, 640, 640);
        wt(fc1_w + (size_t)l * 640 * 2560, fc1wt + (size_t)l * 2560 * 640, 640, 2560);
        wt(fc2_w + (size_t)l * 2560 * 640, fc2wt + (size_t)l * 640 * 2560, 2560, 640);
    }
    wt(lin_w, linwt, 640, 576);

    nsct_fwd<<<kB * kC, 256, 0, stream>>>(x, x3);

    for (int l = 0; l < 2; ++l) {
        int shift = (l == 0) ? 0 : (kWin / 2);
        ln_kernel<true><<<kTok, 256, 0, stream>>>(x3, ln1_g + l * 640, ln1_b + l * 640, bufA, shift);
        gemm_bf16_wmma<0, 0, false><<<dim3(15, 196), 256, 0, stream>>>(
            bufA, qkvwt + (size_t)l * 1920 * 640, qkv_b + l * 1920, nullptr, qkv,
            kTok, 1920, 640, 0);
        attn_wmma<<<dim3(kWins, kHeads), 128, 0, stream>>>(qkv, bufA, shift);
        gemm_bf16_wmma<2, 0, true><<<dim3(5, 196), 256, 0, stream>>>(
            bufA, projwt + (size_t)l * 640 * 640, proj_b + l * 640, x3, nullptr,
            kTok, 640, 640, shift);
        ln_kernel<false><<<kTok, 256, 0, stream>>>(x3, ln2_g + l * 640, ln2_b + l * 640, bufA, 0);
        gemm_bf16_wmma<0, 1, false><<<dim3(20, 196), 256, 0, stream>>>(
            bufA, fc1wt + (size_t)l * 2560 * 640, fc1_b + l * 2560, nullptr, ff,
            kTok, 2560, 640, 0);
        gemm_bf16_wmma<2, 0, false><<<dim3(5, 196), 256, 0, stream>>>(
            ff, fc2wt + (size_t)l * 640 * 2560, fc2_b + l * 640, x3, nullptr,
            kTok, 640, 2560, 0);
    }

    {
        size_t n = (size_t)kTok * 640;
        cast_f32_bf16<<<(unsigned)((n + 255) / 256), 256, 0, stream>>>(x3, bufA, n);
    }
    gemm_bf16_wmma<1, 0, false><<<dim3(5, 196), 256, 0, stream>>>(
        bufA, linwt, lin_b, ybuf, nullptr, kTok, 576, 640, 0);
    ict_out_kernel<<<kTok, 256, 0, stream>>>(ybuf, conv_w, conv_b, (float*)d_out);
}
